// head_24275155157741
// MI455X (gfx1250) — matrix operand order repaired and checked by reading, not compiled, not run on hardware
//
#include <hip/hip_runtime.h>

// Problem constants from the reference
#define BB 4
#define TT 4096
#define CC 768
#define DD 64
#define KT (CC / 32)           // 24 K-tiles in the projection GEMM
#define NCHUNK 32              // split-K chunks over T for K^T V
#define CHUNK (TT / NCHUNK)    // 128 timesteps per chunk

typedef __attribute__((ext_vector_type(16))) _Float16 v16h;
typedef __attribute__((ext_vector_type(8)))  _Float16 v8h;
typedef __attribute__((ext_vector_type(8)))  float    v8f;

// ---------------------------------------------------------------------------
// Fragment mappings used here for v_wmma_f32_16x16x32_f16 (wave32); A/B K order is this code's choice.
// A (16x32, MxK):  lane L<16 -> m=L, K 0..15; lane>=16 -> m=L-16,
//                  K 16..31; 2 packed f16 per VGPR.
// B (32x16, KxN):  lanes 0..15 -> n=L, K 0..15 down 8 VGPRs (2/VGPR);
//                  lanes 16..31 -> n=L-16, K 16..31.
// C/D (16x16 f32): lanes 0..15 -> n=L, VGPR r -> m=r; lanes 16..31 -> m=8+r.
// ---------------------------------------------------------------------------
__device__ __forceinline__ v16h load_A_f32(const float* base, int strideM, int strideK) {
  const int lane = threadIdx.x & 31;
  const int m = lane & 15;
  const int half = lane >> 4;
  v16h a;
#pragma unroll
  for (int e = 0; e < 16; ++e) {
    const int i = e >> 1, j = e & 1;
    const int k = half * 16 + 2 * i + j;
    a[e] = (_Float16)base[m * strideM + k * strideK];
  }
  return a;
}

__device__ __forceinline__ v16h load_A_f16(const _Float16* base, int strideM, int strideK) {
  const int lane = threadIdx.x & 31;
  const int m = lane & 15;
  const int half = lane >> 4;
  v16h a;
#pragma unroll
  for (int e = 0; e < 16; ++e) {
    const int i = e >> 1, j = e & 1;
    const int k = half * 16 + 2 * i + j;
    a[e] = base[m * strideM + k * strideK];
  }
  return a;
}

__device__ __forceinline__ v16h load_B_f16(const _Float16* base, int strideK, int strideN) {
  const int lane = threadIdx.x & 31;
  const int n = lane & 15;
  const int half = lane >> 4;
  v16h bf;
#pragma unroll
  for (int e = 0; e < 16; ++e) {
    const int v = e >> 1, j = e & 1;
    const int k = half * 16 + 2 * v + j;
    bf[e] = base[k * strideK + n * strideN];
  }
  return bf;
}

__device__ __forceinline__ v8f wmma_f16(v16h a, v16h b, v8f c) {
  // (neg_a, A, neg_b, B, c_mod, C, reuse_a, reuse_b)
  return __builtin_amdgcn_wmma_f32_16x16x32_f16(false, a, false, b, (short)0, c,
                                                false, false);
}

// ---------------------------------------------------------------------------
// Kernel 0: pre-pack Wq|Wk|Wv (fp32, [D][C] row-major) into fragment-ready
// f16 B-operands: Wf[mat][ktile][ntile][lane][e].  One wave per fragment;
// runs once per launch so the 1024 projection waves each read a fragment as
// a single aligned 32B v16h load with zero converts.
// ---------------------------------------------------------------------------
__global__ __launch_bounds__(256) void w_pack_kernel(
    const float* __restrict__ Wq, const float* __restrict__ Wk,
    const float* __restrict__ Wv, _Float16* __restrict__ Wf) {
  const int frag = blockIdx.x * 8 + threadIdx.y;  // 0 .. 3*KT*4-1 = 287
  if (frag >= 3 * KT * 4) return;
  const int mat = frag / (KT * 4);
  const int kt = (frag / 4) % KT;
  const int nt = frag % 4;
  const float* W = (mat == 0) ? Wq : (mat == 1) ? Wk : Wv;

  const int lane = threadIdx.x & 31;
  const int n = lane & 15, half = lane >> 4;
  v16h o;
#pragma unroll
  for (int e = 0; e < 16; ++e) {
    const int v = e >> 1, j = e & 1;
    const int k = half * 16 + 2 * v + j;
    // B[k][n] = W[n0+n][k0+k]
    o[e] = (_Float16)W[(size_t)(nt * 16 + n) * CC + kt * 32 + k];
  }
  *(v16h*)(Wf + ((size_t)frag * 32 + lane) * 16) = o;
}

// ---------------------------------------------------------------------------
// Kernel 1: fused QKV projection.  One wave owns a 16-row strip of x (read
// once, fp32->f16 inline) and produces q, k, v.  q stored row-major [t][d]
// (its consumer reads A along contiguous K=d); k and v stored TRANSPOSED
// [b][d][t] so the K^T V kernel gets contiguous-along-K fragments, and the
// store itself vectorizes (8 consecutive-t halves per lane -> one b128).
// ---------------------------------------------------------------------------
__global__ __launch_bounds__(256) void qkv_proj_kernel(
    const float* __restrict__ x, const _Float16* __restrict__ Wf,
    _Float16* __restrict__ qh, _Float16* __restrict__ kT,
    _Float16* __restrict__ vT) {
  const int wid = blockIdx.x * 8 + threadIdx.y;  // 0 .. B*T/16-1 (1024 waves)
  const int row0 = wid * 16;                     // flat row in (B*T, C)
  const int b = row0 / TT;
  const int tloc = row0 % TT;
  const float* xbase = x + (size_t)row0 * CC;
  const int lane = threadIdx.x & 31;

  v8f acc[12] = {};  // [mat][ntile] -> acc[mat*4+ntile]

  for (int kt = 0; kt < KT; ++kt) {
    const v16h a = load_A_f32(xbase + kt * 32, CC, 1);
#pragma unroll
    for (int mat = 0; mat < 3; ++mat)
#pragma unroll
      for (int nt = 0; nt < 4; ++nt) {
        const v16h bf = *(const v16h*)(
            Wf + ((size_t)((mat * KT + kt) * 4 + nt) * 32 + lane) * 16);
        acc[mat * 4 + nt] = wmma_f16(a, bf, acc[mat * 4 + nt]);
      }
  }

  const int n = lane & 15, half = lane >> 4;

  // q: row-major [B*T][D]
#pragma unroll
  for (int nt = 0; nt < 4; ++nt)
#pragma unroll
    for (int r = 0; r < 8; ++r) {
      const int m = half * 8 + r;
      qh[(size_t)(row0 + m) * DD + nt * 16 + n] = (_Float16)acc[nt][r];
    }

  // k, v: transposed [b][d][t]; lane's 8 rows are consecutive t -> b128 store
#pragma unroll
  for (int nt = 0; nt < 4; ++nt) {
    v8h pk, pv;
#pragma unroll
    for (int r = 0; r < 8; ++r) {
      pk[r] = (_Float16)acc[4 + nt][r];
      pv[r] = (_Float16)acc[8 + nt][r];
    }
    const size_t doff = (size_t)(b * DD + nt * 16 + n) * TT + tloc + half * 8;
    *(v8h*)(kT + doff) = pk;
    *(v8h*)(vT + doff) = pv;
  }
}

// ---------------------------------------------------------------------------
// Kernel 2: split-K partials of H = K^T V ([64x64] per batch, contraction
// over T=4096).  Grid = B*NCHUNK blocks x 16 waves; wave (mi,ni) owns one
// 16x16 tile of H for its T-chunk.  With kT/vT in [d][t] layout both A and B
// fragments are contiguous along K -> vectorized b128 loads.
// Deterministic: explicit partials + reduction, no float atomics.
// ---------------------------------------------------------------------------
__global__ __launch_bounds__(512) void ktv_partial_kernel(
    const _Float16* __restrict__ kT, const _Float16* __restrict__ vT,
    float* __restrict__ Hpart) {
  const int b = blockIdx.x / NCHUNK;
  const int chunk = blockIdx.x % NCHUNK;
  const int w = threadIdx.y;  // 0..15
  const int mi = w >> 2, ni = w & 3;
  const int t0 = chunk * CHUNK;

  const _Float16* kbase = kT + ((size_t)(b * DD + mi * 16)) * TT + t0;
  const _Float16* vbase = vT + ((size_t)(b * DD + ni * 16)) * TT + t0;

  v8f acc = {};
#pragma unroll 4
  for (int tt = 0; tt < CHUNK; tt += 32) {
    // A[m][kk] = kT[mi*16+m][t0+tt+kk] : strideM=TT, strideK=1 (contiguous K)
    const v16h a = load_A_f16(kbase + tt, TT, 1);
    // B[kk][n] = vT[ni*16+n][t0+tt+kk] : strideK=1, strideN=TT (contiguous K)
    const v16h bf = load_B_f16(vbase + tt, 1, TT);
    acc = wmma_f16(a, bf, acc);
  }

  const int lane = threadIdx.x & 31;
  const int n = lane & 15, half = lane >> 4;
  float* out = Hpart + ((size_t)chunk * BB + b) * (DD * DD);
#pragma unroll
  for (int r = 0; r < 8; ++r) {
    const int m = mi * 16 + half * 8 + r;
    out[(size_t)m * DD + ni * 16 + n] = acc[r];
  }
}

// ---------------------------------------------------------------------------
// Kernel 3: reduce split-K partials, fold scale = D^-0.5, emit H in f16.
// ---------------------------------------------------------------------------
__global__ void h_reduce_kernel(const float* __restrict__ Hpart,
                                _Float16* __restrict__ Hh) {
  const int idx = blockIdx.x * blockDim.x + threadIdx.x;
  if (idx >= BB * DD * DD) return;
  const int b = idx / (DD * DD);
  const int rem = idx % (DD * DD);
  float s = 0.f;
#pragma unroll
  for (int c = 0; c < NCHUNK; ++c)
    s += Hpart[((size_t)c * BB + b) * (DD * DD) + rem];
  Hh[idx] = (_Float16)(s * 0.125f);  // scale = 64^-0.5
}

// ---------------------------------------------------------------------------
// Kernel 4: out = Q @ H (M=B*T, K=64, N=64).  One wave per 16-row strip,
// 2 K-steps x 4 N-tiles = 8 WMMAs, f32 results straight to d_out.
// ---------------------------------------------------------------------------
__global__ __launch_bounds__(256) void out_kernel(
    const _Float16* __restrict__ qh, const _Float16* __restrict__ Hh,
    float* __restrict__ out) {
  const int wid = blockIdx.x * 8 + threadIdx.y;  // 0..1023
  const int row0 = wid * 16;
  const int b = row0 / TT;  // strip never crosses a batch (T % 16 == 0)
  const _Float16* qbase = qh + (size_t)row0 * DD;
  const _Float16* Hbase = Hh + (size_t)b * DD * DD;

  v8f acc[4] = {};
#pragma unroll
  for (int k0 = 0; k0 < DD; k0 += 32) {
    const v16h a = load_A_f16(qbase + k0, DD, 1);  // contiguous along K=d
#pragma unroll
    for (int nt = 0; nt < 4; ++nt) {
      const v16h bf = load_B_f16(Hbase + (size_t)k0 * DD + nt * 16, DD, 1);
      acc[nt] = wmma_f16(a, bf, acc[nt]);
    }
  }

  const int lane = threadIdx.x & 31;
  const int n = lane & 15, half = lane >> 4;
#pragma unroll
  for (int nt = 0; nt < 4; ++nt)
#pragma unroll
    for (int r = 0; r < 8; ++r) {
      const int m = half * 8 + r;
      out[(size_t)(row0 + m) * DD + nt * 16 + n] = acc[nt][r];
    }
}

// ---------------------------------------------------------------------------
// Workspace layout (needs ~8.6 MB):
//   [0, 2MB)          qh  f16 [B*T][D]        (row-major)
//   [2MB, 4MB)        kT  f16 [B][D][T]       (transposed)
//   [4MB, 6MB)        vT  f16 [B][D][T]       (transposed)
//   [6MB, +288KB)     Wf  f16 fragment-packed weights [3][24][4][32][16]
//   [6.5MB, 8.5MB)    Hpart f32 [NCHUNK][B][64][64]
//   [8.5MB, +32KB)    Hh  f16 [B][64][64]
// ---------------------------------------------------------------------------
extern "C" void kernel_launch(void* const* d_in, const int* in_sizes, int n_in,
                              void* d_out, int out_size, void* d_ws,
                              size_t ws_size, hipStream_t stream) {
  (void)in_sizes; (void)n_in; (void)out_size; (void)ws_size;
  const float* x  = (const float*)d_in[0];
  const float* Wq = (const float*)d_in[1];
  const float* Wk = (const float*)d_in[2];
  const float* Wv = (const float*)d_in[3];
  float* out = (float*)d_out;

  char* ws = (char*)d_ws;
  _Float16* qh = (_Float16*)(ws);
  _Float16* kT = (_Float16*)(ws + (size_t)2 * 1024 * 1024);
  _Float16* vT = (_Float16*)(ws + (size_t)4 * 1024 * 1024);
  _Float16* Wf = (_Float16*)(ws + (size_t)6 * 1024 * 1024);
  float*    Hp = (float*)   (ws + (size_t)6 * 1024 * 1024 + 512 * 1024);
  _Float16* Hh = (_Float16*)(ws + (size_t)8 * 1024 * 1024 + 512 * 1024);

  dim3 blk8(32, 8);    // 8 waves
  dim3 blk16(32, 16);  // 16 waves

  // 0) pre-pack weights into fragment-ready f16 (288 waves)
  w_pack_kernel<<<36, blk8, 0, stream>>>(Wq, Wk, Wv, Wf);
  // 1) fused QKV projection: 1024 waves, x read exactly once
  qkv_proj_kernel<<<128, blk8, 0, stream>>>(x, Wf, qh, kT, vT);
  // 2) split-K K^T V partials: 128 blocks x 16 waves
  ktv_partial_kernel<<<BB * NCHUNK, blk16, 0, stream>>>(kT, vT, Hp);
  // 3) reduce + scale -> Hh (f16)
  h_reduce_kernel<<<(BB * DD * DD + 255) / 256, 256, 0, stream>>>(Hp, Hh);
  // 4) out = Q @ Hh : 1024 waves
  out_kernel<<<128, blk8, 0, stream>>>(qh, Hh, out);
}
